// NonLocal_40707700032191
// MI455X (gfx1250) — matrix operand order repaired and checked by reading, not compiled, not run on hardware
//
#include <hip/hip_runtime.h>

typedef __attribute__((ext_vector_type(16))) __bf16   v16bf;
typedef __attribute__((ext_vector_type(8)))  __bf16   bf16x8;
typedef __attribute__((ext_vector_type(8)))  float    v8f;
typedef __attribute__((ext_vector_type(8)))  unsigned v8u;

namespace {
constexpr int kB = 4;
constexpr int kC = 64;
constexpr int kN = 6400;  // 80*80

static __device__ __forceinline__ __bf16 f2bf(float f) {
  unsigned u = __builtin_bit_cast(unsigned, f);
  u += 0x7FFFu + ((u >> 16) & 1u);  // round-to-nearest-even
  unsigned short h = (unsigned short)(u >> 16);
  return __builtin_bit_cast(__bf16, h);
}

// Pack two floats to packed bf16 (RNE).
static __device__ __forceinline__ unsigned pack_bf(float a, float b) {
  unsigned ua = __builtin_bit_cast(unsigned, a);
  ua += 0x7FFFu + ((ua >> 16) & 1u);
  unsigned ub = __builtin_bit_cast(unsigned, b);
  ub += 0x7FFFu + ((ub >> 16) & 1u);
  return (ua >> 16) | (ub & 0xFFFF0000u);
}

// Exchange value with lane^16 (across the two 16-lane rows of the wave).
static __device__ __forceinline__ unsigned xor16u(unsigned v) {
#if __has_builtin(__builtin_amdgcn_permlanex16)
  return (unsigned)__builtin_amdgcn_permlanex16((int)v, (int)v, (int)0x76543210u,
                                                (int)0xFEDCBA98u, false, false);
#else
  return (unsigned)__shfl_xor((int)v, 16);
#endif
}
static __device__ __forceinline__ float xor16f(float v) {
  return __builtin_bit_cast(float, xor16u(__builtin_bit_cast(unsigned, v)));
}
}  // namespace

// -------------------------------------------------------------------------
// Kernel 1: fused 1x1 projections.
//   theta[B,N,C] bf16, phi[B,N,C] bf16 (contiguous-over-C rows feed WMMA
//   A/B operands directly), g stored transposed as gt[B,C,N] bf16 (rows of
//   G^T feed the second GEMM's A-operand directly).
// -------------------------------------------------------------------------
__global__ void __launch_bounds__(256)
proj_kernel(const float* __restrict__ x,
            const float* __restrict__ w_theta, const float* __restrict__ b_theta,
            const float* __restrict__ w_phi,   const float* __restrict__ b_phi,
            const float* __restrict__ w_g,     const float* __restrict__ b_g,
            __bf16* __restrict__ theta, __bf16* __restrict__ phi,
            __bf16* __restrict__ gt) {
  __shared__ float xs[kC][65];  // +1 pad: column reads hit distinct banks
  const int b  = blockIdx.x / (kN / 64);
  const int n0 = (blockIdx.x % (kN / 64)) * 64;
  const int t  = (int)threadIdx.x;

  const float* xb = x + (size_t)b * kC * kN + n0;
#pragma unroll
  for (int i = 0; i < 16; ++i) {
    int idx = t + 256 * i;  // 64 channels x 64 positions
    xs[idx >> 6][idx & 63] = xb[(size_t)(idx >> 6) * kN + (idx & 63)];
  }
  __syncthreads();

  const int j  = t & 63;   // spatial position within tile
  const int cg = t >> 6;   // channel group 0..3
  for (int i = 0; i < 16; ++i) {
    int c = cg * 16 + i;
    const float* wt = w_theta + c * kC;
    const float* wp = w_phi   + c * kC;
    const float* wg = w_g     + c * kC;
    float st = b_theta[c], sp = b_phi[c], sg = b_g[c];
#pragma unroll
    for (int k = 0; k < kC; ++k) {
      float xv = xs[k][j];
      st += wt[k] * xv;
      sp += wp[k] * xv;
      sg += wg[k] * xv;
    }
    size_t nrow = (size_t)b * kN + n0 + j;
    theta[nrow * kC + c] = f2bf(st);
    phi[nrow * kC + c]   = f2bf(sp);
    gt[((size_t)b * kC + c) * kN + n0 + j] = f2bf(sg);
  }
}

// -------------------------------------------------------------------------
// Kernel 2: flash attention, transposed formulation. One wave per 16-query
// tile.  S^T = phi_tile (A) x theta_tile (B)  -> each lane owns ONE query
// column (N=lq) with 8 key-rows per score tile (M = r + 8*half).
//   * softmax: in-lane tree over 16 values + one permlanex16 (lane^16).
//   * O^T = G^T (A, straight from gt[C,N]) x P^T (B, assembled from the
//     in-register p-values + 8 permlanex16 dword swaps). No LDS at all.
// C/D of O^T: VGPR r <-> channel c = t*16 + r + 8*half, column = query lq.
// -------------------------------------------------------------------------
__global__ void __launch_bounds__(32)
attn_kernel(const __bf16* __restrict__ theta,
            const __bf16* __restrict__ phi,
            const __bf16* __restrict__ gt,
            float* __restrict__ y) {
  const int b    = blockIdx.x / (kN / 16);
  const int q0   = (blockIdx.x % (kN / 16)) * 16;
  const int lane = (int)threadIdx.x;
  const int half = lane >> 4;
  const int lq   = lane & 15;

  // theta B operand (hoisted): lane holds query column q0+lq,
  // K chunks at channel half*8 + {0..7} and half*8 + 16 + {0..7}; two v16bf cover C=64.
  v16bf tB0, tB1;
  {
    const __bf16* trow = theta + ((size_t)b * kN + q0 + lq) * kC + half * 8;
    bf16x8 a0 = *(const bf16x8*)(trow);
    bf16x8 a1 = *(const bf16x8*)(trow + 16);
    bf16x8 a2 = *(const bf16x8*)(trow + 32);
    bf16x8 a3 = *(const bf16x8*)(trow + 48);
#pragma unroll
    for (int e = 0; e < 8; ++e) {
      tB0[e] = a0[e]; tB0[8 + e] = a1[e];
      tB1[e] = a2[e]; tB1[8 + e] = a3[e];
    }
  }

  v8f acc[4];
#pragma unroll
  for (int t = 0; t < 4; ++t) acc[t] = 0.0f;
  float rmax = -3.0e38f;
  float rsum = 0.0f;

  const __bf16* phib = phi + (size_t)b * kN * kC;
  const __bf16* gb   = gt  + (size_t)b * kC * kN;
  const v8f zero = 0.0f;

  for (int kb = 0; kb < kN; kb += 32) {
    // phi A operands: rows = keys kb+lq and kb+16+lq; A-layout K chunks at
    // channel half*8 + {0..7} and half*8 + 16 + {0..7}; k1 adds 32.
    v16bf a0k0, a0k1, a1k0, a1k1;
    {
      const __bf16* pr0 = phib + (size_t)(kb + lq) * kC + half * 8;
      const __bf16* pr1 = phib + (size_t)(kb + 16 + lq) * kC + half * 8;
      bf16x8 u0 = *(const bf16x8*)(pr0);
      bf16x8 u1 = *(const bf16x8*)(pr0 + 16);
      bf16x8 u2 = *(const bf16x8*)(pr0 + 32);
      bf16x8 u3 = *(const bf16x8*)(pr0 + 48);
      bf16x8 w0 = *(const bf16x8*)(pr1);
      bf16x8 w1 = *(const bf16x8*)(pr1 + 16);
      bf16x8 w2 = *(const bf16x8*)(pr1 + 32);
      bf16x8 w3 = *(const bf16x8*)(pr1 + 48);
#pragma unroll
      for (int e = 0; e < 8; ++e) {
        a0k0[e] = u0[e]; a0k0[8 + e] = u1[e];
        a0k1[e] = u2[e]; a0k1[8 + e] = u3[e];
        a1k0[e] = w0[e]; a1k0[8 + e] = w1[e];
        a1k1[e] = w2[e]; a1k1[8 + e] = w3[e];
      }
    }
    // S^T tiles: s0 -> keys kb + r + 8*half, s1 -> keys kb+16 + r + 8*half.
    v8f s0 = __builtin_amdgcn_wmma_f32_16x16x32_bf16(false, a0k0, false, tB0, (short)0, zero, false, false);
    s0     = __builtin_amdgcn_wmma_f32_16x16x32_bf16(false, a0k1, false, tB1, (short)0, s0,   false, false);
    v8f s1 = __builtin_amdgcn_wmma_f32_16x16x32_bf16(false, a1k0, false, tB0, (short)0, zero, false, false);
    s1     = __builtin_amdgcn_wmma_f32_16x16x32_bf16(false, a1k1, false, tB1, (short)0, s1,   false, false);

    // ---- online softmax over this 32-key block (per-lane scalar state) ----
    float m = fmaxf(s0[0], s1[0]);
#pragma unroll
    for (int r = 1; r < 8; ++r) m = fmaxf(m, fmaxf(s0[r], s1[r]));
    m = fmaxf(m, xor16f(m));            // combine the two key-halves
    float nm    = fmaxf(rmax, m);
    float scale = __expf(rmax - nm);
    float p0[8], p1[8];
    float ps = 0.0f;
#pragma unroll
    for (int r = 0; r < 8; ++r) {
      p0[r] = __expf(s0[r] - nm);
      p1[r] = __expf(s1[r] - nm);
      ps += p0[r] + p1[r];
    }
    ps += xor16f(ps);
    rsum = rsum * scale + ps;
    rmax = nm;
#pragma unroll
    for (int t = 0; t < 4; ++t) {
#pragma unroll
      for (int r = 0; r < 8; ++r) acc[t][r] *= scale;
    }

    // ---- assemble P^T B-operand: lane needs keys half*16 + {0..15} for its
    // query. Own registers hold keys {0..7|8..15} (s0) and {16..23|24..31}
    // (s1) depending on half -> swap packed dwords with lane^16.
    v8u pd;
#pragma unroll
    for (int i = 0; i < 4; ++i) {
      unsigned d0 = pack_bf(p0[2 * i], p0[2 * i + 1]);  // own s0 pair
      unsigned d1 = pack_bf(p1[2 * i], p1[2 * i + 1]);  // own s1 pair
      unsigned ax = xor16u(d0);  // other half's s0 (keys 8..15 for half 0)
      unsigned bx = xor16u(d1);  // other half's s1 (keys 16..23 for half 1)
      pd[i]     = half ? bx : d0;  // elements e=0..7  -> K = half*16 + 0..7
      pd[4 + i] = half ? d1 : ax;  // elements e=8..15 -> K = half*16 + 8..15
    }
    v16bf pB = __builtin_bit_cast(v16bf, pd);

    // ---- O^T += G^T x P^T : A from gt rows (channel c = t*16+lq), K
    // at keys kb + half*16 + {0..15} contiguous.
#pragma unroll
    for (int t = 0; t < 4; ++t) {
      const __bf16* gr = gb + (size_t)(t * 16 + lq) * kN + kb + half * 16;
      bf16x8 lo = *(const bf16x8*)(gr);
      bf16x8 hi = *(const bf16x8*)(gr + 8);
      v16bf gA;
#pragma unroll
      for (int e = 0; e < 8; ++e) { gA[e] = lo[e]; gA[8 + e] = hi[e]; }
      acc[t] = __builtin_amdgcn_wmma_f32_16x16x32_bf16(false, gA, false, pB, (short)0, acc[t], false, false);
    }
  }

  // Normalize and write y[B,N,C] f32: lane's query row q0+lq, channels
  // t*16 + 8*half + r are 8 consecutive floats per tile.
  float inv = 1.0f / rsum;
  float* yr = y + ((size_t)b * kN + q0 + lq) * kC + 8 * half;
#pragma unroll
  for (int t = 0; t < 4; ++t) {
#pragma unroll
    for (int r = 0; r < 8; ++r) yr[t * 16 + r] = acc[t][r] * inv;
  }
}

// -------------------------------------------------------------------------
// Kernel 3: W 1x1 conv + deterministic per-block BN partial sums.
// -------------------------------------------------------------------------
__global__ void __launch_bounds__(256)
wconv_kernel(const float* __restrict__ y, const float* __restrict__ w_W,
             const float* __restrict__ b_W, float* __restrict__ wy,
             float* __restrict__ partial) {
  __shared__ float ys[64][65];
  __shared__ float psum[2][kC];
  __shared__ float psq[2][kC];
  const int b  = blockIdx.x / (kN / 64);
  const int n0 = (blockIdx.x % (kN / 64)) * 64;
  const int t  = (int)threadIdx.x;

  const float* yb = y + ((size_t)b * kN + n0) * kC;
#pragma unroll
  for (int i = 0; i < 16; ++i) {
    int idx = t + 256 * i;
    ys[idx >> 6][idx & 63] = yb[idx];  // ys[pos][channel]
  }
  __syncthreads();

  const int j    = t & 63;
  const int cg   = t >> 6;
  const int lane = t & 31;
  const int wv   = (t >> 5) & 1;
  for (int i = 0; i < 16; ++i) {
    int c = cg * 16 + i;
    const float* wr = w_W + c * kC;
    float s = b_W[c];
#pragma unroll
    for (int k = 0; k < kC; ++k) s += wr[k] * ys[j][k];
    wy[((size_t)b * kC + c) * kN + n0 + j] = s;
    float s1 = s, s2 = s * s;
#pragma unroll
    for (int m = 16; m >= 1; m >>= 1) { s1 += __shfl_xor(s1, m); s2 += __shfl_xor(s2, m); }
    if (lane == 0) { psum[wv][c] = s1; psq[wv][c] = s2; }
  }
  __syncthreads();
  if (t < kC) {
    partial[(size_t)blockIdx.x * 128 + t]      = psum[0][t] + psum[1][t];
    partial[(size_t)blockIdx.x * 128 + 64 + t] = psq[0][t] + psq[1][t];
  }
}

// Kernel 4: fold 400 block-partials into per-channel sum / sumsq.
__global__ void __launch_bounds__(128)
stats_reduce_kernel(const float* __restrict__ partial, float* __restrict__ stats) {
  int t = (int)threadIdx.x;  // 0..127
  float s = 0.0f;
  for (int blk = 0; blk < kB * (kN / 64); ++blk) s += partial[(size_t)blk * 128 + t];
  stats[t] = s;
}

// Kernel 5: BN (training-mode batch stats) + affine + residual.
__global__ void __launch_bounds__(256)
bn_kernel(const float* __restrict__ wy, const float* __restrict__ x,
          const float* __restrict__ gamma, const float* __restrict__ beta,
          const float* __restrict__ stats, float* __restrict__ out) {
  size_t i = (size_t)blockIdx.x * 256 + threadIdx.x;
  int c = (int)((i / kN) % kC);
  const float inv = 1.0f / (float)(kB * kN);
  float mean = stats[c] * inv;
  float var  = fmaxf(stats[64 + c] * inv - mean * mean, 0.0f);
  float r = rsqrtf(var + 1e-5f);
  out[i] = (wy[i] - mean) * r * gamma[c] + beta[c] + x[i];
}

// -------------------------------------------------------------------------
extern "C" void kernel_launch(void* const* d_in, const int* in_sizes, int n_in,
                              void* d_out, int out_size, void* d_ws, size_t ws_size,
                              hipStream_t stream) {
  (void)in_sizes; (void)n_in; (void)out_size; (void)ws_size;
  const float* x       = (const float*)d_in[0];
  const float* w_theta = (const float*)d_in[1];
  const float* b_theta = (const float*)d_in[2];
  const float* w_phi   = (const float*)d_in[3];
  const float* b_phi   = (const float*)d_in[4];
  const float* w_g     = (const float*)d_in[5];
  const float* b_g     = (const float*)d_in[6];
  const float* w_W     = (const float*)d_in[7];
  const float* b_W     = (const float*)d_in[8];
  const float* gamma   = (const float*)d_in[9];
  const float* beta    = (const float*)d_in[10];
  float* out = (float*)d_out;

  // Workspace layout (all 256B aligned): ~23.1 MB total.
  char* ws = (char*)d_ws;
  __bf16* theta = (__bf16*)(ws + 0);         // B*N*C bf16 = 3,276,800
  __bf16* phi   = (__bf16*)(ws + 3276800);   // B*N*C bf16
  __bf16* gt    = (__bf16*)(ws + 6553600);   // B*C*N bf16
  float*  y     = (float*)(ws + 9830400);    // B*N*C f32 = 6,553,600
  float*  wy    = (float*)(ws + 16384000);   // B*C*N f32
  float*  part  = (float*)(ws + 22937600);   // 400*128 f32 = 204,800
  float*  stats = (float*)(ws + 23142400);   // 128 f32

  proj_kernel<<<kB * (kN / 64), 256, 0, stream>>>(x, w_theta, b_theta, w_phi, b_phi,
                                                  w_g, b_g, theta, phi, gt);
  attn_kernel<<<kB * (kN / 16), 32, 0, stream>>>(theta, phi, gt, y);
  wconv_kernel<<<kB * (kN / 64), 256, 0, stream>>>(y, w_W, b_W, wy, part);
  stats_reduce_kernel<<<1, 128, 0, stream>>>(part, stats);
  bn_kernel<<<(unsigned)((size_t)kB * kC * kN / 256), 256, 0, stream>>>(wy, x, gamma, beta, stats, out);
}
